// TreeLSTM_17154099380582
// MI455X (gfx1250) — compile-verified, matrix operand order checked by reading, not run on hardware
//
#include <hip/hip_runtime.h>
#include <hip/hip_bf16.h>
#include <math.h>
#include <stdint.h>

// ---------------------------------------------------------------------------
// CDNA5 (gfx1250) TreeLSTM pipeline.
// wave32; v_wmma_f32_16x16x32_bf16; global_load_async_to_lds_b128 staging.
// ---------------------------------------------------------------------------

typedef __bf16 v16bf __attribute__((ext_vector_type(16)));
typedef float  v8f   __attribute__((ext_vector_type(8)));

union Frag { v16bf v; uint4 q[2]; };

__device__ __forceinline__ float sigf(float x) { return 1.0f / (1.0f + __expf(-x)); }

// 16x32 bf16 fragment (A layout; also serves B when weight is row-major [N,K]).
// lane: idx=lane&15 row; lhalf=lane>>4 selects K-halves per CDNA5 VGPR layout.
__device__ __forceinline__ v16bf load_frag(const __bf16* __restrict__ row, int kb, int lhalf) {
  Frag f;
  f.q[0] = *(const uint4*)(row + kb + lhalf * 8);
  f.q[1] = *(const uint4*)(row + kb + 16 + lhalf * 8);
  return f.v;
}

// Async copy of one 16-byte chunk global -> LDS (gfx1250 VFLAT async encoding).
__device__ __forceinline__ void async_b128(unsigned lds_addr, const void* gaddr) {
  asm volatile("global_load_async_to_lds_b128 %0, %1, off"
               :: "v"(lds_addr), "v"(gaddr) : "memory");
}
__device__ __forceinline__ void async_wait0() {
  asm volatile("s_wait_asynccnt 0" ::: "memory");
}

__device__ __forceinline__ void store_tile(v8f acc, float* __restrict__ out,
                                           __bf16* __restrict__ out_bf,
                                           const float* __restrict__ aux,
                                           int ldo, int m0, int n0,
                                           int lidx, int lhalf, int act) {
#pragma unroll
  for (int r = 0; r < 8; ++r) {
    const int row = m0 + r + lhalf * 8;
    const int col = n0 + lidx;
    float v = acc[r];
    if (act == 1) v = fmaxf(v, 0.0f);
    else if (act == 2) v = sigf(v) * aux[(size_t)row * ldo + col];
    out[(size_t)row * ldo + col] = v;
    if (out_bf) out_bf[(size_t)row * ldo + col] = (__bf16)v;
  }
}

// ---------------------------------------------------------------------------
// NT GEMM: out[m][n] = act( sum_k A[m][k]*W[n][k] + bias[n] )
// Block 256 thr = 8 waves; wave tile 32x32 (4 independent WMMA chains);
// block tile 64(M) x 128(N).  act: 0 none, 1 relu, 2 sigmoid(x)*aux.
// ---------------------------------------------------------------------------
__global__ __launch_bounds__(256) void gemm_nt_kernel(
    const __bf16* __restrict__ A, int lda,
    const __bf16* __restrict__ W, int ldw,
    const float* __restrict__ bias,
    float* __restrict__ out, int ldo,
    __bf16* __restrict__ out_bf,
    const float* __restrict__ aux,
    int K, int act)
{
  const int wave  = threadIdx.x >> 5;
  const int lane  = threadIdx.x & 31;
  const int lidx  = lane & 15;
  const int lhalf = lane >> 4;
  const int m0 = blockIdx.x * 64 + (wave >> 2) * 32;
  const int n0 = blockIdx.y * 128 + (wave & 3) * 32;

  const float b0 = bias ? bias[n0 + lidx] : 0.0f;
  const float b1 = bias ? bias[n0 + 16 + lidx] : 0.0f;
  v8f acc00, acc01, acc10, acc11;
#pragma unroll
  for (int r = 0; r < 8; ++r) { acc00[r] = b0; acc01[r] = b1; acc10[r] = b0; acc11[r] = b1; }

  const __bf16* arow0 = A + (size_t)(m0 + lidx) * lda;
  const __bf16* arow1 = A + (size_t)(m0 + 16 + lidx) * lda;
  const __bf16* wrow0 = W + (size_t)(n0 + lidx) * ldw;
  const __bf16* wrow1 = W + (size_t)(n0 + 16 + lidx) * ldw;

  for (int kb = 0; kb < K; kb += 32) {
    __builtin_prefetch(wrow0 + kb + 128, 0, 1);
    __builtin_prefetch(wrow1 + kb + 128, 0, 1);
    v16bf a0 = load_frag(arow0, kb, lhalf);
    v16bf a1 = load_frag(arow1, kb, lhalf);
    v16bf w0 = load_frag(wrow0, kb, lhalf);
    v16bf w1 = load_frag(wrow1, kb, lhalf);
    acc00 = __builtin_amdgcn_wmma_f32_16x16x32_bf16(false, a0, false, w0, (short)0, acc00, false, false);
    acc01 = __builtin_amdgcn_wmma_f32_16x16x32_bf16(false, a0, false, w1, (short)0, acc01, false, false);
    acc10 = __builtin_amdgcn_wmma_f32_16x16x32_bf16(false, a1, false, w0, (short)0, acc10, false, false);
    acc11 = __builtin_amdgcn_wmma_f32_16x16x32_bf16(false, a1, false, w1, (short)0, acc11, false, false);
  }

  store_tile(acc00, out, out_bf, aux, ldo, m0,      n0,      lidx, lhalf, act);
  store_tile(acc01, out, out_bf, aux, ldo, m0,      n0 + 16, lidx, lhalf, act);
  store_tile(acc10, out, out_bf, aux, ldo, m0 + 16, n0,      lidx, lhalf, act);
  store_tile(acc11, out, out_bf, aux, ldo, m0 + 16, n0 + 16, lidx, lhalf, act);
}

// ---------------------------------------------------------------------------
// NN GEMM (B row-major [K,N]) for the small head GEMM z1 = relu(target @ xm).
// Block 256 = 8 waves, each one 16x16 tile; block tile 32x64.
// ---------------------------------------------------------------------------
__global__ __launch_bounds__(256) void gemm_nn_kernel(
    const __bf16* __restrict__ A, int lda,
    const __bf16* __restrict__ B, int ldb,
    float* __restrict__ out, int ldo,
    __bf16* __restrict__ out_bf,
    int K)
{
  const int wave  = threadIdx.x >> 5;
  const int lane  = threadIdx.x & 31;
  const int lidx  = lane & 15;
  const int lhalf = lane >> 4;
  const int m0 = blockIdx.x * 32 + (wave >> 2) * 16;
  const int n0 = blockIdx.y * 64 + (wave & 3) * 16;

  v8f acc;
#pragma unroll
  for (int r = 0; r < 8; ++r) acc[r] = 0.0f;

  const __bf16* arow = A + (size_t)(m0 + lidx) * lda;
  for (int kb = 0; kb < K; kb += 32) {
    v16bf av = load_frag(arow, kb, lhalf);
    Frag fb;
#pragma unroll
    for (int j = 0; j < 8; ++j) {
      fb.v[j]     = B[(size_t)(kb + lhalf * 8 + j) * ldb + n0 + lidx];
      fb.v[8 + j] = B[(size_t)(kb + 16 + lhalf * 8 + j) * ldb + n0 + lidx];
    }
    acc = __builtin_amdgcn_wmma_f32_16x16x32_bf16(false, av, false, fb.v, (short)0, acc, false, false);
  }
  store_tile(acc, out, out_bf, nullptr, ldo, m0, n0, lidx, lhalf, /*relu*/1);
}

// ---------------------------------------------------------------------------
// One LSTM timestep (one direction). 4 waves = gates i,f,g,o for the same
// 16(B)x16(H) tile. The shared A operand [x_t | h_prev] (16 x (320+512)) is
// staged once per block into LDS with global_load_async_to_lds_b128, then all
// 4 waves read it via ds_load_b128. LDS row strides padded (328/520 elems,
// 16B aligned) to spread bank starts.
// ---------------------------------------------------------------------------
__global__ __launch_bounds__(128) void lstm_step_kernel(
    const __bf16* __restrict__ xin,   // [128,320]
    const __bf16* __restrict__ hprev, // [128,512]
    const float*  __restrict__ cprev, // [128,512]
    const __bf16* __restrict__ wih,   // [2048,320]
    const __bf16* __restrict__ whh,   // [2048,512]
    const float*  __restrict__ bih,
    const float*  __restrict__ bhh,
    __bf16* __restrict__ hnext,       // [128,512]
    float*  __restrict__ cnext,       // [128,512]
    __bf16* __restrict__ xcat)        // pre-offset; row stride 1024
{
  __shared__ __bf16 xs[16 * 328];
  __shared__ __bf16 hs[16 * 520];
  __shared__ float  ls[4][16][16];

  const int gate  = threadIdx.x >> 5;       // 0=i 1=f 2=g 3=o
  const int lane  = threadIdx.x & 31;
  const int lidx  = lane & 15;
  const int lhalf = lane >> 4;
  const int m0 = blockIdx.x * 16;           // batch tile
  const int n0 = blockIdx.y * 16;           // hidden tile
  const int wbase = gate * 512 + n0;

  // ---- async stage A into LDS ----
  const unsigned xs_base = (unsigned)(uintptr_t)&xs[0];
  const unsigned hs_base = (unsigned)(uintptr_t)&hs[0];
  for (int q = threadIdx.x; q < 640; q += 128) {          // 16 rows x 40 chunks
    const int r = q / 40, j = q - r * 40;
    async_b128(xs_base + (unsigned)(r * 328 + j * 8) * 2,
               xin + (size_t)(m0 + r) * 320 + j * 8);
  }
  for (int q = threadIdx.x; q < 1024; q += 128) {         // 16 rows x 64 chunks
    const int r = q >> 6, j = q & 63;
    async_b128(hs_base + (unsigned)(r * 520 + j * 8) * 2,
               hprev + (size_t)(m0 + r) * 512 + j * 8);
  }
  async_wait0();
  __syncthreads();

  const float binit = bih[wbase + lidx] + bhh[wbase + lidx];
  v8f accA, accB;                                          // 2 independent chains
#pragma unroll
  for (int r = 0; r < 8; ++r) { accA[r] = binit; accB[r] = 0.0f; }

  const __bf16* axrow = xs + lidx * 328;
  const __bf16* ahrow = hs + lidx * 520;
  const __bf16* wxrow = wih + (size_t)(wbase + lidx) * 320;
  const __bf16* whrow = whh + (size_t)(wbase + lidx) * 512;

  for (int kb = 0; kb < 320; kb += 64) {                   // 5 pairs... 320 = 5*64
    accA = __builtin_amdgcn_wmma_f32_16x16x32_bf16(
        false, load_frag(axrow, kb, lhalf), false, load_frag(wxrow, kb, lhalf),
        (short)0, accA, false, false);
    if (kb + 32 < 320)
      accB = __builtin_amdgcn_wmma_f32_16x16x32_bf16(
          false, load_frag(axrow, kb + 32, lhalf), false, load_frag(wxrow, kb + 32, lhalf),
          (short)0, accB, false, false);
  }
  for (int kb = 0; kb < 512; kb += 64) {
    accA = __builtin_amdgcn_wmma_f32_16x16x32_bf16(
        false, load_frag(ahrow, kb, lhalf), false, load_frag(whrow, kb, lhalf),
        (short)0, accA, false, false);
    accB = __builtin_amdgcn_wmma_f32_16x16x32_bf16(
        false, load_frag(ahrow, kb + 32, lhalf), false, load_frag(whrow, kb + 32, lhalf),
        (short)0, accB, false, false);
  }

#pragma unroll
  for (int r = 0; r < 8; ++r) ls[gate][r + lhalf * 8][lidx] = accA[r] + accB[r];
  __syncthreads();

  for (int e = threadIdx.x; e < 256; e += 128) {
    const int m = e >> 4, n = e & 15;
    const float gi = ls[0][m][n], gf = ls[1][m][n], gg = ls[2][m][n], go = ls[3][m][n];
    const size_t idx = (size_t)(m0 + m) * 512 + n0 + n;
    const float c = sigf(gf) * cprev[idx] + sigf(gi) * tanhf(gg);
    const float h = sigf(go) * tanhf(c);
    cnext[idx] = c;
    hnext[idx] = (__bf16)h;
    xcat[(size_t)(m0 + m) * 1024 + n0 + n] = (__bf16)h;
  }
}

// ---------------------------------------------------------------------------
// TreeLSTM node update for one level. 3 waves = gates i,o,u; shared htild tile
// staged into LDS via async loads (when the level has children).
// ---------------------------------------------------------------------------
__global__ __launch_bounds__(96) void tree_gates_kernel(
    const __bf16* __restrict__ htild,  // [M,512]
    const __bf16* __restrict__ uiou,   // [1536,512]
    const float* __restrict__ biou,    // [1536]
    const float* __restrict__ iou0,    // pre-offset, ld 1536
    const float* __restrict__ cred,    // [M,512] or nullptr
    float*  __restrict__ cout,         // pre-offset, ld 512
    __bf16* __restrict__ hout,         // pre-offset, ld 512
    int has_children)
{
  __shared__ __bf16 hsx[16 * 520];
  __shared__ float  ls[3][16][16];

  const int gate  = threadIdx.x >> 5;  // 0=i 1=o 2=u
  const int lane  = threadIdx.x & 31;
  const int lidx  = lane & 15;
  const int lhalf = lane >> 4;
  const int m0 = blockIdx.x * 16;
  const int n0 = blockIdx.y * 16;
  const int colbase = gate * 512 + n0;

  const float binit = biou[colbase + lidx];
  v8f accA, accB;
#pragma unroll
  for (int r = 0; r < 8; ++r) { accA[r] = binit; accB[r] = 0.0f; }

  if (has_children) {
    const unsigned hs_base = (unsigned)(uintptr_t)&hsx[0];
    for (int q = threadIdx.x; q < 1024; q += 96) {
      const int r = q >> 6, j = q & 63;
      async_b128(hs_base + (unsigned)(r * 520 + j * 8) * 2,
                 htild + (size_t)(m0 + r) * 512 + j * 8);
    }
    async_wait0();
    __syncthreads();

    const __bf16* arow = hsx + lidx * 520;
    const __bf16* wrow = uiou + (size_t)(colbase + lidx) * 512;
    for (int kb = 0; kb < 512; kb += 64) {
      accA = __builtin_amdgcn_wmma_f32_16x16x32_bf16(
          false, load_frag(arow, kb, lhalf), false, load_frag(wrow, kb, lhalf),
          (short)0, accA, false, false);
      accB = __builtin_amdgcn_wmma_f32_16x16x32_bf16(
          false, load_frag(arow, kb + 32, lhalf), false, load_frag(wrow, kb + 32, lhalf),
          (short)0, accB, false, false);
    }
  }
#pragma unroll
  for (int r = 0; r < 8; ++r)
    accA[r] += accB[r] + iou0[(size_t)(m0 + r + lhalf * 8) * 1536 + colbase + lidx];

#pragma unroll
  for (int r = 0; r < 8; ++r) ls[gate][r + lhalf * 8][lidx] = accA[r];
  __syncthreads();

  for (int e = threadIdx.x; e < 256; e += 96) {
    const int m = e >> 4, n = e & 15;
    const float gi = ls[0][m][n], go = ls[1][m][n], gu = ls[2][m][n];
    const size_t idx = (size_t)(m0 + m) * 512 + n0 + n;
    const float cr = cred ? cred[idx] : 0.0f;
    const float c = sigf(gi) * tanhf(gu) + cr;
    const float h = sigf(go) * tanhf(c);
    cout[idx] = c;
    hout[idx] = (__bf16)h;
  }
}

// Pairwise child combine over the binary heap.
__global__ void tree_combine_kernel(
    const __bf16* __restrict__ htree, const float* __restrict__ fcbuf,
    __bf16* __restrict__ htild, float* __restrict__ cred,
    int tok0, size_t total)
{
  const size_t stride = (size_t)gridDim.x * blockDim.x;
  for (size_t i = (size_t)blockIdx.x * blockDim.x + threadIdx.x; i < total; i += stride) {
    const int p = (int)(i >> 9);
    const int k = (int)(i & 511);
    const int t = tok0 + (p >> 7);
    const int b = p & 127;
    const int c1 = 2 * t + 1, c2 = 2 * t + 2;
    float sh = 0.0f, sc = 0.0f;
    if (c1 < 64) {
      const size_t r = ((size_t)c1 * 128 + b) * 512 + k;
      sh += (float)htree[r]; sc += fcbuf[r];
    }
    if (c2 < 64) {
      const size_t r = ((size_t)c2 * 128 + b) * 512 + k;
      sh += (float)htree[r]; sc += fcbuf[r];
    }
    htild[i] = (__bf16)sh;
    cred[i]  = sc;
  }
}

// fp32 -> bf16 with K padding.
__global__ void cvt_pad_bf16_kernel(const float* __restrict__ src, __bf16* __restrict__ dst,
                                    int rows, int sc, int dc)
{
  const size_t total = (size_t)rows * dc;
  const size_t stride = (size_t)gridDim.x * blockDim.x;
  for (size_t i = (size_t)blockIdx.x * blockDim.x + threadIdx.x; i < total; i += stride) {
    const int r = (int)(i / dc), k = (int)(i % dc);
    dst[i] = (k < sc) ? (__bf16)src[(size_t)r * sc + k] : (__bf16)0.0f;
  }
}

// Embedding gather -> bf16, E padded 300 -> 320.
__global__ void embed_kernel(const float* __restrict__ table, const int* __restrict__ ids,
                             __bf16* __restrict__ dst)
{
  const size_t total = (size_t)8192 * 320;
  const size_t stride = (size_t)gridDim.x * blockDim.x;
  for (size_t i = (size_t)blockIdx.x * blockDim.x + threadIdx.x; i < total; i += stride) {
    const int r = (int)(i / 320), k = (int)(i % 320);
    const int id = ids[r];
    dst[i] = (k < 300) ? (__bf16)table[(size_t)id * 300 + k] : (__bf16)0.0f;
  }
}

__global__ void zero_kernel(float* __restrict__ p, size_t nwords)
{
  const size_t stride = (size_t)gridDim.x * blockDim.x;
  for (size_t i = (size_t)blockIdx.x * blockDim.x + threadIdx.x; i < nwords; i += stride)
    p[i] = 0.0f;
}

// out = relu(z2 @ lin2_w^T + b)  [128,3]
__global__ void lin2_kernel(const float* __restrict__ z2, const float* __restrict__ w,
                            const float* __restrict__ b, float* __restrict__ out)
{
  const int o = blockIdx.x * blockDim.x + threadIdx.x;
  if (o >= 384) return;
  const int row = o / 3, cls = o % 3;
  float s = b[cls];
  for (int k = 0; k < 512; ++k) s += z2[row * 512 + k] * w[cls * 512 + k];
  out[o] = fmaxf(s, 0.0f);
}

// ---------------------------------------------------------------------------
extern "C" void kernel_launch(void* const* d_in, const int* in_sizes, int n_in,
                              void* d_out, int out_size, void* d_ws, size_t ws_size,
                              hipStream_t stream) {
  (void)in_sizes; (void)n_in; (void)out_size; (void)ws_size;
  const float* embed_table = (const float*)d_in[0];
  const float* w_ih_f = (const float*)d_in[1];
  const float* w_hh_f = (const float*)d_in[2];
  const float* b_ih_f = (const float*)d_in[3];
  const float* b_hh_f = (const float*)d_in[4];
  const float* w_ih_b = (const float*)d_in[5];
  const float* w_hh_b = (const float*)d_in[6];
  const float* b_ih_b = (const float*)d_in[7];
  const float* b_hh_b = (const float*)d_in[8];
  const float* w_iou  = (const float*)d_in[9];
  const float* u_iou  = (const float*)d_in[10];
  const float* b_iou  = (const float*)d_in[11];
  const float* u_f_w  = (const float*)d_in[12];
  const float* u_f_b  = (const float*)d_in[13];
  const float* lin1_w = (const float*)d_in[14];
  const float* lin1_b = (const float*)d_in[15];
  const float* lin2_w = (const float*)d_in[16];
  const float* lin2_b = (const float*)d_in[17];
  const float* target = (const float*)d_in[18];
  const int*   ids    = (const int*)d_in[19];
  float* out = (float*)d_out;

  size_t off = 0;
  auto alloc = [&](size_t bytes) -> void* {
    void* p = (char*)d_ws + off;
    off += (bytes + 255) & ~(size_t)255;
    return p;
  };
  __bf16* emb_bf   = (__bf16*)alloc((size_t)8192 * 320 * 2);
  __bf16* wihf_bf  = (__bf16*)alloc((size_t)2048 * 320 * 2);
  __bf16* wihb_bf  = (__bf16*)alloc((size_t)2048 * 320 * 2);
  __bf16* whhf_bf  = (__bf16*)alloc((size_t)2048 * 512 * 2);
  __bf16* whhb_bf  = (__bf16*)alloc((size_t)2048 * 512 * 2);
  __bf16* wiou_bf  = (__bf16*)alloc((size_t)1536 * 1024 * 2);
  __bf16* uiou_bf  = (__bf16*)alloc((size_t)1536 * 512 * 2);
  __bf16* ufw_bf   = (__bf16*)alloc((size_t)512 * 512 * 2);
  __bf16* lin1_bf  = (__bf16*)alloc((size_t)512 * 512 * 2);
  __bf16* targ_bf  = (__bf16*)alloc((size_t)128 * 1024 * 2);
  __bf16* x_bf     = (__bf16*)alloc((size_t)8192 * 1024 * 2);
  __bf16* htree_bf = (__bf16*)alloc((size_t)8192 * 512 * 2);
  float*  ctree    = (float*)alloc((size_t)8192 * 512 * 4);
  float*  iou0     = (float*)alloc((size_t)8192 * 1536 * 4);
  float*  fcbuf    = (float*)alloc((size_t)8192 * 512 * 4);
  __bf16* htild_bf = (__bf16*)alloc((size_t)4096 * 512 * 2);
  float*  credbuf  = (float*)alloc((size_t)4096 * 512 * 4);
  __bf16* hstate   = (__bf16*)alloc((size_t)4 * 128 * 512 * 2); // [dir*2+buf]
  float*  cstate   = (float*)alloc((size_t)4 * 128 * 512 * 4);
  float*  z1       = (float*)alloc((size_t)128 * 512 * 4);
  __bf16* z1_bf    = (__bf16*)alloc((size_t)128 * 512 * 2);
  float*  z2       = (float*)alloc((size_t)128 * 512 * 4);

  const int ZT = 256;
  auto nblk = [](size_t total, int thr) { return (int)((total + thr - 1) / thr); };

  // ---- Prep: bf16 weight conversions + embedding gather + zero states ----
  cvt_pad_bf16_kernel<<<nblk((size_t)2048 * 320, ZT), ZT, 0, stream>>>(w_ih_f, wihf_bf, 2048, 300, 320);
  cvt_pad_bf16_kernel<<<nblk((size_t)2048 * 320, ZT), ZT, 0, stream>>>(w_ih_b, wihb_bf, 2048, 300, 320);
  cvt_pad_bf16_kernel<<<nblk((size_t)2048 * 512, ZT), ZT, 0, stream>>>(w_hh_f, whhf_bf, 2048, 512, 512);
  cvt_pad_bf16_kernel<<<nblk((size_t)2048 * 512, ZT), ZT, 0, stream>>>(w_hh_b, whhb_bf, 2048, 512, 512);
  cvt_pad_bf16_kernel<<<nblk((size_t)1536 * 1024, ZT), ZT, 0, stream>>>(w_iou, wiou_bf, 1536, 1024, 1024);
  cvt_pad_bf16_kernel<<<nblk((size_t)1536 * 512, ZT), ZT, 0, stream>>>(u_iou, uiou_bf, 1536, 512, 512);
  cvt_pad_bf16_kernel<<<nblk((size_t)512 * 512, ZT), ZT, 0, stream>>>(u_f_w, ufw_bf, 512, 512, 512);
  cvt_pad_bf16_kernel<<<nblk((size_t)512 * 512, ZT), ZT, 0, stream>>>(lin1_w, lin1_bf, 512, 512, 512);
  cvt_pad_bf16_kernel<<<nblk((size_t)128 * 1024, ZT), ZT, 0, stream>>>(target, targ_bf, 128, 1024, 1024);
  embed_kernel<<<nblk((size_t)8192 * 320, ZT), ZT, 0, stream>>>(embed_table, ids, emb_bf);
  zero_kernel<<<nblk((size_t)4 * 128 * 512 * 2 / 4, ZT), ZT, 0, stream>>>((float*)hstate, (size_t)4 * 128 * 512 * 2 / 4);
  zero_kernel<<<nblk((size_t)4 * 128 * 512, ZT), ZT, 0, stream>>>(cstate, (size_t)4 * 128 * 512);

  // ---- Bidirectional LSTM: 64 sequential steps, 2 directions ----
  const size_t HS = (size_t)128 * 512;
  for (int t = 0; t < 64; ++t) {
    const int cur = t & 1, nxt = cur ^ 1;
    lstm_step_kernel<<<dim3(8, 32), 128, 0, stream>>>(
        emb_bf + (size_t)t * 128 * 320,
        hstate + (0 * 2 + cur) * HS, cstate + (0 * 2 + cur) * HS,
        wihf_bf, whhf_bf, b_ih_f, b_hh_f,
        hstate + (0 * 2 + nxt) * HS, cstate + (0 * 2 + nxt) * HS,
        x_bf + (size_t)t * 128 * 1024);
    const int tb = 63 - t;
    lstm_step_kernel<<<dim3(8, 32), 128, 0, stream>>>(
        emb_bf + (size_t)tb * 128 * 320,
        hstate + (1 * 2 + cur) * HS, cstate + (1 * 2 + cur) * HS,
        wihb_bf, whhb_bf, b_ih_b, b_hh_b,
        hstate + (1 * 2 + nxt) * HS, cstate + (1 * 2 + nxt) * HS,
        x_bf + (size_t)tb * 128 * 1024 + 512);
  }

  // ---- iou0 = x @ w_iou^T : [8192,1024] x [1536,1024]^T ----
  gemm_nt_kernel<<<dim3(8192 / 64, 1536 / 128), 256, 0, stream>>>(
      x_bf, 1024, wiou_bf, 1024, nullptr, iou0, 1536, nullptr, nullptr, 1024, 0);

  // ---- Tree propagation: levels 6 (leaf-most) down to 0 (roots) ----
  for (int lvl = 6; lvl >= 0; --lvl) {
    const int tok0 = (1 << lvl) - 1;
    const int tok1 = (((1 << (lvl + 1)) - 1) < 64) ? ((1 << (lvl + 1)) - 1) : 64;
    const int ntok = tok1 - tok0;
    const int ctok0 = (1 << (lvl + 1)) - 1;
    const int ctok1 = (((1 << (lvl + 2)) - 1) < 64) ? ((1 << (lvl + 2)) - 1) : 64;
    const int nctok = (ctok1 > ctok0) ? (ctok1 - ctok0) : 0;

    if (nctok > 0) {
      const size_t coff = (size_t)ctok0 * 128 * 512;
      const int Mc = nctok * 128;
      gemm_nt_kernel<<<dim3(Mc / 64, 512 / 128), 256, 0, stream>>>(
          htree_bf + coff, 512, ufw_bf, 512, u_f_b,
          fcbuf + coff, 512, nullptr, ctree + coff, 512, 2);
      const size_t totp = (size_t)ntok * 128 * 512;
      tree_combine_kernel<<<nblk(totp, ZT), ZT, 0, stream>>>(
          htree_bf, fcbuf, htild_bf, credbuf, tok0, totp);
    }
    const size_t poff = (size_t)tok0 * 128 * 512;
    tree_gates_kernel<<<dim3(ntok * 128 / 16, 32), 96, 0, stream>>>(
        htild_bf, uiou_bf, b_iou,
        iou0 + (size_t)tok0 * 128 * 1536,
        (nctok > 0) ? credbuf : nullptr,
        ctree + poff, htree_bf + poff,
        nctok > 0 ? 1 : 0);
  }

  // ---- Head ----
  gemm_nn_kernel<<<dim3(128 / 32, 512 / 64), 256, 0, stream>>>(
      targ_bf, 1024, htree_bf, 512, z1, 512, z1_bf, 1024);
  gemm_nt_kernel<<<dim3(128 / 64, 512 / 128), 256, 0, stream>>>(
      z1_bf, 512, lin1_bf, 512, lin1_b, z2, 512, nullptr, nullptr, 512, 1);
  lin2_kernel<<<2, 256, 0, stream>>>(z2, lin2_w, lin2_b, out);
}
